// HGNN_67800353735002
// MI455X (gfx1250) — matrix operand order repaired and checked by reading, not compiled, not run on hardware
//
#include <hip/hip_runtime.h>

#define N_NODES 50000
#define F_DIM   64
#define G_DIM   64
#define C_CAT   4
#define NNZ_E   800000
#define L0_DIM  128
#define EMB_DIM 64
#define CG      (C_CAT * G_DIM)   // 256
#define BN_EPS  1e-5f

typedef _Float16 h16;
typedef __attribute__((ext_vector_type(16))) _Float16 v16h;
typedef __attribute__((ext_vector_type(8)))  float    v8f;

__device__ __forceinline__ v8f wmma16(v16h a, v16h b, v8f c) {
  return __builtin_amdgcn_wmma_f32_16x16x32_f16(
      false, a, false, b, (short)0, c, false, false);
}

// ---- A fragment 16x32 (MxK) from row-major f32, convert to f16 -----------
__device__ __forceinline__ v16h load_a_f32(const float* __restrict__ src, int ld, int lane) {
  int m = lane & 15, hh = lane >> 4;
  const float* p = src + (size_t)m * ld + hh * 8;
  v16h a;
#pragma unroll
  for (int i = 0; i < 8; ++i) a[i] = (h16)p[i];
#pragma unroll
  for (int i = 0; i < 8; ++i) a[8 + i] = (h16)p[16 + i];
  return a;
}

// ---- A fragment 16x32 from row-major f16 (no conversion, pure b128) ------
__device__ __forceinline__ v16h load_a_f16(const h16* __restrict__ src, int ld, int lane) {
  int m = lane & 15, hh = lane >> 4;
  const h16* p = src + (size_t)m * ld + hh * 8;
  v16h a;
#pragma unroll
  for (int i = 0; i < 8; ++i) a[i] = p[i];
#pragma unroll
  for (int i = 0; i < 8; ++i) a[8 + i] = p[16 + i];
  return a;
}

// A loader with fused BatchNorm (x*scale[k]+shift[k]) + ELU -> f16
__device__ __forceinline__ v16h load_a_bn_elu(const float* __restrict__ src, int ld, int k0,
                                              const float* __restrict__ scv,
                                              const float* __restrict__ shv, int lane) {
  int m = lane & 15, hh = lane >> 4;
  const float* p = src + (size_t)m * ld + k0 + hh * 8;
  v16h a;
#pragma unroll
  for (int i = 0; i < 8; ++i) {
    int k = k0 + hh * 8 + i;
    float v = fmaf(p[i], scv[k], shv[k]);
    v = (v > 0.f) ? v : expm1f(v);
    a[i] = (h16)v;
  }
#pragma unroll
  for (int i = 0; i < 8; ++i) {
    int k = k0 + 16 + hh * 8 + i;
    float v = fmaf(p[16 + i], scv[k], shv[k]);
    v = (v > 0.f) ? v : expm1f(v);
    a[8 + i] = (h16)v;
  }
  return a;
}

// B fragment fetch from pre-swizzled f16 weights (32 contiguous B/lane)
__device__ __forceinline__ v16h load_b_frag(const h16* __restrict__ Bf, int NT,
                                            int kc, int nt, int lane) {
  return *(const v16h*)(Bf + (size_t)(((kc * NT + nt) * 32) + lane) * 16);
}

// ---- weight pre-swizzle: f32 [K,N] row-major -> f16 fragment order -------
__device__ __forceinline__ void conv_frag(const float* __restrict__ src,
                                          h16* __restrict__ dst, int Ncols, int t) {
  int e = t & 15;
  int lane = (t >> 4) & 31;
  int frag = t >> 9;
  int NT = Ncols >> 4;
  int nt = frag % NT, kc = frag / NT;
  int k = kc * 32 + (e >> 3) * 16 + (lane >> 4) * 8 + (e & 7);
  int n = nt * 16 + (lane & 15);
  dst[t] = (h16)src[(size_t)k * Ncols + n];
}

__global__ void conv_weights_kernel(const float* __restrict__ Ws,
                                    const float* __restrict__ W0,
                                    const float* __restrict__ W1,
                                    h16* __restrict__ Wsf,
                                    h16* __restrict__ W0f,
                                    h16* __restrict__ W1f) {
  const int SZ_WS = C_CAT * F_DIM * G_DIM;   // 16384
  const int SZ_W0 = CG * L0_DIM;             // 32768
  const int SZ_W1 = L0_DIM * EMB_DIM;        // 8192
  int t = blockIdx.x * blockDim.x + threadIdx.x;
  if (t < SZ_WS) {
    int c = t / (F_DIM * G_DIM), r = t % (F_DIM * G_DIM);
    conv_frag(Ws + (size_t)c * F_DIM * G_DIM, Wsf + (size_t)c * F_DIM * G_DIM, G_DIM, r);
  } else if (t < SZ_WS + SZ_W0) {
    conv_frag(W0, W0f, L0_DIM, t - SZ_WS);
  } else if (t < SZ_WS + SZ_W0 + SZ_W1) {
    conv_frag(W1, W1f, EMB_DIM, t - SZ_WS - SZ_W0);
  }
}

// ---- kernel 1: XW[c] = X @ Ws[c] -> f16 ; one wave = 16 rows x 64 cols ---
__global__ void gemm_xw_kernel(const float* __restrict__ X,
                               const h16* __restrict__ Wsf,
                               h16* __restrict__ XW) {
  int wave = (int)((blockIdx.x * blockDim.x + threadIdx.x) >> 5);
  const int MT = N_NODES / 16;
  if (wave >= C_CAT * MT) return;              // wave-uniform
  int lane = threadIdx.x & 31;
  int c = wave / MT, mt = wave % MT;
  const float* Arow = X + (size_t)(mt * 16) * F_DIM;
  const h16* Bf = Wsf + (size_t)c * F_DIM * G_DIM;
  const int NT = G_DIM / 16;                   // 4
  v8f acc[4] = {};
#pragma unroll
  for (int kc = 0; kc < F_DIM / 32; ++kc) {    // 2
    v16h a = load_a_f32(Arow + kc * 32, F_DIM, lane);
#pragma unroll
    for (int nt = 0; nt < 4; ++nt)
      acc[nt] = wmma16(a, load_b_frag(Bf, NT, kc, nt, lane), acc[nt]);
  }
  int n = lane & 15, hh = lane >> 4;
  h16* D = XW + ((size_t)c * N_NODES + mt * 16) * G_DIM;
#pragma unroll
  for (int nt = 0; nt < 4; ++nt)
#pragma unroll
    for (int i = 0; i < 8; ++i)
      D[(size_t)(i + 8 * hh) * G_DIM + nt * 16 + n] = (h16)acc[nt][i];
}

// ---- kernel 2: edge scatter: H[dst, c*64+g] += XW[c, src, g] (pk f16) ----
// One wave per edge: 32 lanes x 1 dword gather (coalesced 128B/row) and one
// GLOBAL_ATOMIC_PK_ADD_F16 per lane -> 102M packed atomics, L2-resident.
__global__ void edge_agg_kernel(const long long* __restrict__ E,
                                const h16* __restrict__ XW,
                                h16* __restrict__ Hc) {
  long long tid = (long long)blockIdx.x * blockDim.x + threadIdx.x;
  long long e = tid >> 5;
  if (e >= (long long)C_CAT * NNZ_E) return;
  int g2 = ((int)tid & 31) * 2;
  int c = (int)(e / NNZ_E);
  long long k = e - (long long)c * NNZ_E;
  long long dst = E[((size_t)c * 2) * NNZ_E + k];
  long long src = E[((size_t)c * 2 + 1) * NNZ_E + k];
  unsigned int val = *(const unsigned int*)(XW + ((size_t)c * N_NODES + (size_t)src) * G_DIM + g2);
  h16* d = Hc + (size_t)dst * CG + c * G_DIM + g2;
  asm volatile("global_atomic_pk_add_f16 %0, %1, off"
               : : "v"(d), "v"(val) : "memory");
}

// ---- kernel 3: h = Hcat(f16) @ W0 + b0 ; one wave = 16 rows x 64 cols ----
__global__ void gemm_h0_kernel(const h16* __restrict__ Hc,
                               const h16* __restrict__ W0f,
                               const float* __restrict__ b0,
                               float* __restrict__ hbuf) {
  int wave = (int)((blockIdx.x * blockDim.x + threadIdx.x) >> 5);
  const int MT = N_NODES / 16;
  if (wave >= MT * 2) return;
  int lane = threadIdx.x & 31;
  int mt = wave >> 1, nth = (wave & 1) * 4;
  const h16* Arow = Hc + (size_t)(mt * 16) * CG;
  const int NT = L0_DIM / 16;                  // 8
  v8f acc[4] = {};
#pragma unroll
  for (int kc = 0; kc < CG / 32; ++kc) {       // 8
    v16h a = load_a_f16(Arow + kc * 32, CG, lane);
#pragma unroll
    for (int j = 0; j < 4; ++j)
      acc[j] = wmma16(a, load_b_frag(W0f, NT, kc, nth + j, lane), acc[j]);
  }
  int n = lane & 15, hh = lane >> 4;
  float* D = hbuf + (size_t)(mt * 16) * L0_DIM;
#pragma unroll
  for (int j = 0; j < 4; ++j) {
    float bias = b0[(nth + j) * 16 + n];
#pragma unroll
    for (int i = 0; i < 8; ++i)
      D[(size_t)(i + 8 * hh) * L0_DIM + (nth + j) * 16 + n] = acc[j][i] + bias;
  }
}

// ---- kernel 4a: per-column sum / sumsq of h ------------------------------
__global__ void bn_stats_kernel(const float* __restrict__ h, float* __restrict__ stats) {
  int col = threadIdx.x & (L0_DIM - 1);
  int rpb = blockDim.x / L0_DIM;
  int row = blockIdx.x * rpb + (threadIdx.x >> 7);
  int stride = gridDim.x * rpb;
  float s = 0.f, s2 = 0.f;
  for (int r = row; r < N_NODES; r += stride) {
    float v = h[(size_t)r * L0_DIM + col];
    s += v; s2 += v * v;
  }
  atomicAdd(&stats[col], s);
  atomicAdd(&stats[L0_DIM + col], s2);
}

// ---- kernel 4b: stats -> per-column scale/shift --------------------------
__global__ void bn_final_kernel(const float* __restrict__ stats,
                                const float* __restrict__ gamma,
                                const float* __restrict__ beta,
                                float* __restrict__ scv, float* __restrict__ shv) {
  int c = threadIdx.x;
  float inv_n = 1.0f / (float)N_NODES;
  float mean = stats[c] * inv_n;
  float var  = stats[L0_DIM + c] * inv_n - mean * mean;
  float s = gamma[c] * rsqrtf(var + BN_EPS);
  scv[c] = s;
  shv[c] = beta[c] - mean * s;
}

// ---- kernel 5: out = elu(bn(h)) @ W1 + b1 ; one wave = 16 x 64 -----------
__global__ void gemm_out_kernel(const float* __restrict__ hbuf,
                                const float* __restrict__ scv,
                                const float* __restrict__ shv,
                                const h16* __restrict__ W1f,
                                const float* __restrict__ b1,
                                float* __restrict__ out) {
  int wave = (int)((blockIdx.x * blockDim.x + threadIdx.x) >> 5);
  const int MT = N_NODES / 16;
  if (wave >= MT) return;
  int lane = threadIdx.x & 31;
  int mt = wave;
  const float* Arow = hbuf + (size_t)(mt * 16) * L0_DIM;
  const int NT = EMB_DIM / 16;                 // 4
  v8f acc[4] = {};
#pragma unroll
  for (int kc = 0; kc < L0_DIM / 32; ++kc) {   // 4
    v16h a = load_a_bn_elu(Arow, L0_DIM, kc * 32, scv, shv, lane);
#pragma unroll
    for (int nt = 0; nt < 4; ++nt)
      acc[nt] = wmma16(a, load_b_frag(W1f, NT, kc, nt, lane), acc[nt]);
  }
  int n = lane & 15, hh = lane >> 4;
  float* D = out + (size_t)(mt * 16) * EMB_DIM;
#pragma unroll
  for (int nt = 0; nt < 4; ++nt) {
    float bias = b1[nt * 16 + n];
#pragma unroll
    for (int i = 0; i < 8; ++i)
      D[(size_t)(i + 8 * hh) * EMB_DIM + nt * 16 + n] = acc[nt][i] + bias;
  }
}

extern "C" void kernel_launch(void* const* d_in, const int* in_sizes, int n_in,
                              void* d_out, int out_size, void* d_ws, size_t ws_size,
                              hipStream_t stream) {
  const long long* A  = (const long long*)d_in[0];   // [C,2,NNZ] int64
  const float* X      = (const float*)d_in[1];       // [N,F]
  const float* Ws     = (const float*)d_in[2];       // [C,F,G]
  const float* W0     = (const float*)d_in[3];       // [CG,L0]
  const float* b0     = (const float*)d_in[4];       // [L0]
  const float* gamma  = (const float*)d_in[5];       // [L0]
  const float* beta   = (const float*)d_in[6];       // [L0]
  const float* W1     = (const float*)d_in[7];       // [L0,EMB]
  const float* b1     = (const float*)d_in[8];       // [EMB]
  float* out = (float*)d_out;                        // [N,EMB] f32

  // workspace layout: f32 region first, then f16 region (all 32B-aligned)
  float* hbuf  = (float*)d_ws;                              // N*L0 f32
  float* stats = hbuf + (size_t)N_NODES * L0_DIM;           // 2*L0
  float* scv   = stats + 2 * L0_DIM;
  float* shv   = scv + L0_DIM;
  h16* XWh = (h16*)(shv + L0_DIM);                          // C*N*G halves
  h16* Hch = XWh + (size_t)C_CAT * N_NODES * G_DIM;         // N*CG halves
  h16* Wsf = Hch + (size_t)N_NODES * CG;                    // C*F*G halves
  h16* W0f = Wsf + (size_t)C_CAT * F_DIM * G_DIM;           // CG*L0
  h16* W1f = W0f + (size_t)CG * L0_DIM;                     // L0*EMB

  // zero accumulation buffers every call (graph-capturable)
  hipMemsetAsync(Hch, 0, (size_t)N_NODES * CG * sizeof(h16), stream);
  hipMemsetAsync(stats, 0, 2 * L0_DIM * sizeof(float), stream);

  // 0) one-shot weight f16 pre-swizzle into B-fragment order
  {
    int total = C_CAT * F_DIM * G_DIM + CG * L0_DIM + L0_DIM * EMB_DIM;  // 57344
    conv_weights_kernel<<<(total + 255) / 256, 256, 0, stream>>>(Ws, W0, W1, Wsf, W0f, W1f);
  }
  // 1) per-category projection (WMMA, 16x64 tile/wave, f16 output)
  {
    int waves = C_CAT * (N_NODES / 16);                     // 12500
    gemm_xw_kernel<<<(waves * 32 + 255) / 256, 256, 0, stream>>>(X, Wsf, XWh);
  }
  // 2) edge scatter-sum (one wave per edge, packed f16 atomics)
  {
    long long threads = (long long)C_CAT * NNZ_E * 32;      // 102.4M
    edge_agg_kernel<<<(int)((threads + 255) / 256), 256, 0, stream>>>(A, XWh, Hch);
  }
  // 3) Linear0 (WMMA, f16 A loads, 16x64 tile/wave)
  {
    int waves = (N_NODES / 16) * 2;                         // 6250
    gemm_h0_kernel<<<(waves * 32 + 255) / 256, 256, 0, stream>>>(Hch, W0f, b0, hbuf);
  }
  // 4) batchnorm statistics
  bn_stats_kernel<<<1024, 256, 0, stream>>>(hbuf, stats);
  bn_final_kernel<<<1, L0_DIM, 0, stream>>>(stats, gamma, beta, scv, shv);
  // 5) fused BN+ELU+Linear1 (WMMA, 16x64 tile/wave)
  {
    int waves = N_NODES / 16;                               // 3125
    gemm_out_kernel<<<(waves * 32 + 255) / 256, 256, 0, stream>>>(hbuf, scv, shv, W1f, b1, out);
  }
}
